// AttentionHead_41334765256968
// MI455X (gfx1250) — matrix operand order repaired and checked by reading, not compiled, not run on hardware
//
#include <hip/hip_runtime.h>
#include <hip/hip_bf16.h>

typedef __attribute__((ext_vector_type(16))) _Float16 v16h;
typedef __attribute__((ext_vector_type(8)))  _Float16 half8;
typedef __attribute__((ext_vector_type(8)))  float    v8f;
typedef __attribute__((ext_vector_type(4)))  unsigned int u32x4;
typedef __attribute__((ext_vector_type(8)))  int          i32x8;
typedef __attribute__((ext_vector_type(4)))  int          i32x4;

#define NE   1024
#define HD   64
#define SEQ  2048
#define NBAT 8
#define ROWS (NBAT * SEQ)

static __device__ __forceinline__ v8f wmma_f16(v16h a, v16h b, v8f c) {
    return __builtin_amdgcn_wmma_f32_16x16x32_f16(false, a, false, b, (short)0, c,
                                                  false, false);
}

static __device__ __forceinline__ v16h hcat(half8 lo, half8 hi) {
    return __builtin_shufflevector(lo, hi, 0,1,2,3,4,5,6,7,8,9,10,11,12,13,14,15);
}

// A-fragment (16x32 f16) from f16 row.  half j -> K = j+16*hh.
static __device__ __forceinline__ v16h load_a_frag(const _Float16* rowPtr, int hh) {
    half8 lo = *(const half8*)(rowPtr + 16 * hh);
    half8 hi = *(const half8*)(rowPtr + 16 * hh + 8);
    return hcat(lo, hi);
}

// A-fragment from an f32 row in LDS, converting to f16 in registers.
static __device__ __forceinline__ v16h load_a_frag_f32(const float* rowPtr, int hh) {
    const float4 c0 = *(const float4*)(rowPtr + 16 * hh);
    const float4 c1 = *(const float4*)(rowPtr + 16 * hh + 4);
    const float4 c2 = *(const float4*)(rowPtr + 16 * hh + 8);
    const float4 c3 = *(const float4*)(rowPtr + 16 * hh + 12);
    v16h a;
    a[0]  = (_Float16)c0.x; a[1]  = (_Float16)c0.y;
    a[2]  = (_Float16)c0.z; a[3]  = (_Float16)c0.w;
    a[4]  = (_Float16)c1.x; a[5]  = (_Float16)c1.y;
    a[6]  = (_Float16)c1.z; a[7]  = (_Float16)c1.w;
    a[8]  = (_Float16)c2.x; a[9]  = (_Float16)c2.y;
    a[10] = (_Float16)c2.z; a[11] = (_Float16)c2.w;
    a[12] = (_Float16)c3.x; a[13] = (_Float16)c3.y;
    a[14] = (_Float16)c3.z; a[15] = (_Float16)c3.w;
    return a;
}

// ---------------------------------------------------------------------------
// Tensor Data Mover: 2D tile load, tensor declared exactly as the tile
// (always in-bounds), dim0 row stride in elements.  D# bitfields per
// CDNA5 ISA.  data_size_log: 0=1B,1=2B,2=4B.
// Tracked by TENSORcnt; completion via s_wait_tensorcnt.
// This toolchain's builtin takes 6 args (g0 x4, g1 x8, g2 x4, g3 x4, x8, cpol).
// ---------------------------------------------------------------------------
static __device__ __forceinline__ void tdm_load_2d(unsigned lds_addr,
                                                   const void* gaddr,
                                                   unsigned tile_d0, unsigned tile_d1,
                                                   unsigned long long stride_d0,
                                                   unsigned data_size_log) {
    const unsigned long long ga = (unsigned long long)(size_t)gaddr;
    u32x4 g0;
    g0[0] = 1u;                                   // count=1, user descriptor
    g0[1] = lds_addr;                             // LDS byte address of tile
    g0[2] = (unsigned)ga;                         // global_addr[31:0]
    g0[3] = (unsigned)((ga >> 32) & 0x01FFFFFFull) | (2u << 30);  // [56:32]|type=2
    i32x8 g1;
    g1[0] = (int)(data_size_log << 16);           // wg_mask=0, data_size, no pad
    g1[1] = (int)((tile_d0 & 0xFFFFu) << 16);     // tensor_dim0[15:0]
    g1[2] = (int)(((tile_d0 >> 16) & 0xFFFFu) | ((tile_d1 & 0xFFFFu) << 16));
    g1[3] = (int)(((tile_d1 >> 16) & 0xFFFFu) | ((tile_d0 & 0xFFFFu) << 16)); // tile_dim0
    g1[4] = (int)(tile_d1 & 0xFFFFu);             // tile_dim1; tile_dim2=0
    g1[5] = (int)(stride_d0 & 0xFFFFFFFFull);     // tensor_dim0_stride[31:0]
    g1[6] = (int)((stride_d0 >> 32) & 0xFFFFull); // stride[47:32]; dim1_stride=0
    g1[7] = 0;
    const i32x4 gz4 = {0, 0, 0, 0};               // groups 2/3: dims 0 (unused)
    const i32x8 gz8 = {0, 0, 0, 0, 0, 0, 0, 0};
    __builtin_amdgcn_tensor_load_to_lds(g0, g1, gz4, gz4, gz8, 0);
}

// ---------------------------------------------------------------------------
// Kernel 0 (runs once): transpose Wq/Wk/Wv to f16 wt3[3*64][1024] so the QKV
// B-operand is a contiguous 2D tile for the TDM.
// ---------------------------------------------------------------------------
__global__ __launch_bounds__(256) void wtrans_kernel(
    const float* __restrict__ Wq, const float* __restrict__ Wk,
    const float* __restrict__ Wv, _Float16* __restrict__ wt3)
{
    const int row = blockIdx.x;                // 0..191: w*64 + n
    const int w = row >> 6, n = row & 63;
    const float* W = (w == 0) ? Wq : ((w == 1) ? Wk : Wv);
#pragma unroll
    for (int p = 0; p < 4; ++p) {
        const int kk = p * 256 + threadIdx.x;
        wt3[(size_t)row * NE + kk] = (_Float16)W[(size_t)kk * HD + n];
    }
}

// ---------------------------------------------------------------------------
// Kernel 1: fused QKV projection with TDM double-buffered staging.
// x[16384,1024] f32 -> q,k [16384,64] f16, v transposed vT[64,16384] f16.
// ---------------------------------------------------------------------------
__global__ __launch_bounds__(256) void qkv_kernel(
    const float* __restrict__ x, const _Float16* __restrict__ wt3,
    _Float16* __restrict__ qo, _Float16* __restrict__ ko,
    _Float16* __restrict__ vTo)
{
    __shared__ __attribute__((aligned(128))) float    xs[2][128 * 32];  // 2x16KB
    __shared__ __attribute__((aligned(128))) _Float16 ws[2][192 * 32];  // 2x12KB

    const int t    = threadIdx.x;
    const int wave = t >> 5;
    const int lane = t & 31;
    const int n    = lane & 15;
    const int hh   = lane >> 4;
    const int r0   = blockIdx.x * 128;

    v8f z = {};
    v8f acc[3][4];
#pragma unroll
    for (int w = 0; w < 3; ++w)
#pragma unroll
        for (int nt = 0; nt < 4; ++nt) acc[w][nt] = z;

    const int NK = NE / 32;
    if (wave == 0) {
        tdm_load_2d((unsigned)(size_t)&xs[0][0], x + (size_t)r0 * NE, 32, 128, NE, 2);
        tdm_load_2d((unsigned)(size_t)&ws[0][0], wt3, 32, 192, NE, 1);
    }

    for (int i = 0; i < NK; ++i) {
        const int buf = i & 1;
        if (wave == 0) __builtin_amdgcn_s_wait_tensorcnt(0);
        __syncthreads();  // publish TDM-written tile; prior buffer reads done
        if (wave == 0 && i + 1 < NK) {
            const int k0 = (i + 1) * 32;
            tdm_load_2d((unsigned)(size_t)&xs[buf ^ 1][0],
                        x + (size_t)r0 * NE + k0, 32, 128, NE, 2);
            tdm_load_2d((unsigned)(size_t)&ws[buf ^ 1][0],
                        wt3 + k0, 32, 192, NE, 1);
        }

        const v16h a = load_a_frag_f32(&xs[buf][(wave * 16 + n) * 32], hh);
#pragma unroll
        for (int w = 0; w < 3; ++w) {
#pragma unroll
            for (int nt = 0; nt < 4; ++nt) {
                // B layout: half j -> K = j + 16*hh, column N = lane%16.
                const v16h b =
                    *(const v16h*)&ws[buf][(w * 64 + nt * 16 + n) * 32 + 16 * hh];
                acc[w][nt] = wmma_f16(a, b, acc[w][nt]);
            }
        }
    }

    // C layout: reg r, lane -> row = r + 8*hh, col = nt*16 + lane%16.
#pragma unroll
    for (int nt = 0; nt < 4; ++nt) {
#pragma unroll
        for (int r = 0; r < 8; ++r) {
            const int grow = r0 + wave * 16 + r + 8 * hh;
            const int gcol = nt * 16 + n;
            qo[(size_t)grow * HD + gcol]    = (_Float16)acc[0][nt][r];
            ko[(size_t)grow * HD + gcol]    = (_Float16)acc[1][nt][r];
            vTo[(size_t)gcol * ROWS + grow] = (_Float16)acc[2][nt][r];
        }
    }
}

// ---------------------------------------------------------------------------
// Kernel 2: flash attention, online softmax, TDM double-buffered k/vT tiles.
// One block = 128 query rows of one batch; each wave owns 16 rows.
// ---------------------------------------------------------------------------
__global__ __launch_bounds__(256) void attn_kernel(
    const _Float16* __restrict__ q, const _Float16* __restrict__ k,
    const _Float16* __restrict__ vT, const int* __restrict__ zmask,
    float* __restrict__ out)
{
    __shared__ __attribute__((aligned(128))) _Float16 kt[2][64 * 64];   // [key][dim]
    __shared__ __attribute__((aligned(128))) _Float16 vtT[2][64 * 64];  // [dim][key]
    __shared__ __attribute__((aligned(128))) _Float16 pt[8][16 * 64];   // per-wave P

    const int t    = threadIdx.x;
    const int wave = t >> 5;
    const int lane = t & 31;
    const int n    = lane & 15;
    const int hh   = lane >> 4;
    const int b    = blockIdx.y;
    const int qb   = blockIdx.x;
    const int qr0  = qb * 128 + wave * 16;
    const size_t rowbase = (size_t)b * SEQ;

    // q fragments for this wave's 16 rows (A row m = lane%16), 2 K-chunks.
    v16h qf0, qf1;
    {
        const _Float16* rp = q + (rowbase + qr0 + n) * HD;
        qf0 = load_a_frag(rp, hh);
        qf1 = load_a_frag(rp + 32, hh);
    }

    v8f z = {};
    v8f o[4];
#pragma unroll
    for (int nt = 0; nt < 4; ++nt) o[nt] = z;

    float mrow[8], lrow[8];
#pragma unroll
    for (int r = 0; r < 8; ++r) { mrow[r] = -1e30f; lrow[r] = 0.0f; }

    const float scale = 0.022097086912079608f;  // 2048^-0.5 (ref scales by C^-0.5)
    const int nkb = 2 * (qb + 1);               // causal: keys <= qb*128+127

    if (wave == 0) {
        tdm_load_2d((unsigned)(size_t)&kt[0][0],  k + rowbase * HD, 64, 64, HD, 1);
        tdm_load_2d((unsigned)(size_t)&vtT[0][0], vT + rowbase, 64, 64, ROWS, 1);
    }

    for (int jb = 0; jb < nkb; ++jb) {
        const int buf = jb & 1;
        const int j0  = jb * 64;
        if (wave == 0) __builtin_amdgcn_s_wait_tensorcnt(0);
        __syncthreads();
        if (wave == 0 && jb + 1 < nkb) {
            const int j0n = (jb + 1) * 64;
            tdm_load_2d((unsigned)(size_t)&kt[buf ^ 1][0],
                        k + (rowbase + j0n) * HD, 64, 64, HD, 1);
            tdm_load_2d((unsigned)(size_t)&vtT[buf ^ 1][0],
                        vT + rowbase + j0n, 64, 64, ROWS, 1);
        }

        // S = q . k^T   (k rows in LDS serve directly as the B operand)
        v8f s[4];
#pragma unroll
        for (int nt = 0; nt < 4; ++nt) s[nt] = z;
#pragma unroll
        for (int c = 0; c < 2; ++c) {
            const v16h af = c ? qf1 : qf0;
#pragma unroll
            for (int nt = 0; nt < 4; ++nt) {
                const v16h bf =
                    *(const v16h*)&kt[buf][(nt * 16 + n) * 64 + c * 32 + 16 * hh];
                s[nt] = wmma_f16(af, bf, s[nt]);
            }
        }

        // scale + causal mask in C layout
#pragma unroll
        for (int nt = 0; nt < 4; ++nt)
#pragma unroll
            for (int r = 0; r < 8; ++r) {
                const int gq = qr0 + r + 8 * hh;
                const int gk = j0 + nt * 16 + n;
                const float vsc = s[nt][r] * scale;
                s[nt][r] = (gk > gq) ? -1e30f : vsc;
            }

        // online softmax: row max / rescale / exp / row sum
        float mnew[8], alpha[8];
#pragma unroll
        for (int r = 0; r < 8; ++r) {
            float mx = fmaxf(fmaxf(s[0][r], s[1][r]), fmaxf(s[2][r], s[3][r]));
#pragma unroll
            for (int d = 1; d < 16; d <<= 1) mx = fmaxf(mx, __shfl_xor(mx, d, 32));
            mnew[r]  = fmaxf(mrow[r], mx);
            alpha[r] = __expf(mrow[r] - mnew[r]);
            mrow[r]  = mnew[r];
        }
        float rs[8];
#pragma unroll
        for (int r = 0; r < 8; ++r) rs[r] = 0.0f;
#pragma unroll
        for (int nt = 0; nt < 4; ++nt)
#pragma unroll
            for (int r = 0; r < 8; ++r) {
                const float p = __expf(s[nt][r] - mnew[r]);
                s[nt][r] = p;
                rs[r] += p;
            }
#pragma unroll
        for (int r = 0; r < 8; ++r) {
            float sm = rs[r];
#pragma unroll
            for (int d = 1; d < 16; d <<= 1) sm += __shfl_xor(sm, d, 32);
            lrow[r] = lrow[r] * alpha[r] + sm;
        }
#pragma unroll
        for (int nt = 0; nt < 4; ++nt)
#pragma unroll
            for (int r = 0; r < 8; ++r) o[nt][r] *= alpha[r];

        // Stage P (C layout -> row-major) so it can be re-read as A fragments.
        _Float16* pw = &pt[wave][0];
#pragma unroll
        for (int nt = 0; nt < 4; ++nt)
#pragma unroll
            for (int r = 0; r < 8; ++r)
                pw[(r + 8 * hh) * 64 + nt * 16 + n] = (_Float16)s[nt][r];
        __syncthreads();

        // O += P @ V  (vT rows in LDS serve directly as the B operand)
        const v16h pf0 = load_a_frag(pw + n * 64, hh);
        const v16h pf1 = load_a_frag(pw + n * 64 + 32, hh);
#pragma unroll
        for (int c = 0; c < 2; ++c) {
            const v16h af = c ? pf1 : pf0;
#pragma unroll
            for (int nt = 0; nt < 4; ++nt) {
                const v16h bf =
                    *(const v16h*)&vtT[buf][(nt * 16 + n) * 64 + c * 32 + 16 * hh];
                o[nt] = wmma_f16(af, bf, o[nt]);
            }
        }
    }

    // Epilogue: divide by softmax sum, apply post-softmax zero mask, store f32.
#pragma unroll
    for (int nt = 0; nt < 4; ++nt)
#pragma unroll
        for (int r = 0; r < 8; ++r) {
            const int gq = qr0 + r + 8 * hh;
            const float val =
                (zmask[rowbase + gq] != 0) ? 0.0f : (o[nt][r] / lrow[r]);
            out[(rowbase + gq) * HD + nt * 16 + n] = val;
        }
}

// ---------------------------------------------------------------------------
extern "C" void kernel_launch(void* const* d_in, const int* in_sizes, int n_in,
                              void* d_out, int out_size, void* d_ws, size_t ws_size,
                              hipStream_t stream) {
    const float* x     = (const float*)d_in[0];
    const float* Wq    = (const float*)d_in[1];
    const float* Wk    = (const float*)d_in[2];
    const float* Wv    = (const float*)d_in[3];
    const int*   zmask = (const int*)d_in[4];
    float*       out   = (float*)d_out;

    _Float16* qws  = (_Float16*)d_ws;                 // [16384][64]
    _Float16* kws  = qws + (size_t)ROWS * HD;         // [16384][64]
    _Float16* vTws = kws + (size_t)ROWS * HD;         // [64][16384]
    _Float16* wt3  = vTws + (size_t)ROWS * HD;        // [192][1024]

    wtrans_kernel<<<dim3(192), 256, 0, stream>>>(Wq, Wk, Wv, wt3);
    qkv_kernel<<<dim3(ROWS / 128), 256, 0, stream>>>(x, wt3, qws, kws, vTws);
    attn_kernel<<<dim3(SEQ / 128, NBAT), 256, 0, stream>>>(qws, kws, vTws, zmask, out);
}
